// EncoderLayer_61357902791037
// MI455X (gfx1250) — matrix operand order repaired and checked by reading, not compiled, not run on hardware
//
#include <hip/hip_runtime.h>
#include <hip/hip_bf16.h>

#define B_ 2
#define S_ 512
#define H_ 256
#define F_ 1024
#define BS_ (B_ * S_)

typedef __attribute__((ext_vector_type(16))) __bf16 v16bf;
typedef __attribute__((ext_vector_type(8)))  __bf16 v8bf;
typedef __attribute__((ext_vector_type(8)))  float  v8f;

// ---------------------------------------------------------------------------
// CDNA5 has a hardware TANH trans op (v_tanh_f32). Use the builtin when the
// toolchain exposes it; otherwise a 2-op exp-based identity that saturates
// correctly at +/-inf.
// ---------------------------------------------------------------------------
__device__ __forceinline__ float fast_tanh(float x) {
#if __has_builtin(__builtin_amdgcn_tanhf)
  return __builtin_amdgcn_tanhf(x);
#elif __has_builtin(__builtin_amdgcn_tanh_f32)
  return __builtin_amdgcn_tanh_f32(x);
#else
  float e = __expf(2.0f * x);
  return 1.0f - 2.0f / (e + 1.0f);
#endif
}

// ---------------------------------------------------------------------------
// 32x32 output tile per wave via 2x2 register blocking of 16x16x32 bf16 WMMA.
//   A  : 32 x K row-major (lda)  -- lane = M row inside each 16-row fragment
//   Bt : 32 x K row-major (ldb)  -- B stored transposed, lane = N row
// Fragment layouts (ISA 7.12.2):
//   A frag: lane r holds K = {hf*8..+7} in v[0..3], {16+hf*8..+7} in v[4..7]
//   B frag: same K layout as A frag: lane n holds K = {hf*8..+7} in v[0..3],
//           {16+hf*8..+7} in v[4..7] -> two 16B reads per lane per step.
// 4 WMMAs per K-step from 8 b128 loads (2 loads/WMMA vs 4 unblocked).
// ---------------------------------------------------------------------------
__device__ __forceinline__ void wmma_tile32(const __bf16* __restrict__ A, int lda,
                                            const __bf16* __restrict__ Bt, int ldb,
                                            int K,
                                            v8f& c00, v8f& c01, v8f& c10, v8f& c11) {
  const int lane = threadIdx.x & 31;
  const int r  = lane & 15;
  const int hf = lane >> 4;
  const __bf16* ap0 = A  + (size_t)r * lda;
  const __bf16* ap1 = A  + (size_t)(16 + r) * lda;
  const __bf16* bp0 = Bt + (size_t)r * ldb;
  const __bf16* bp1 = Bt + (size_t)(16 + r) * ldb;
  for (int k0 = 0; k0 < K; k0 += 32) {
    union { v16bf v; v8bf h[2]; } a0, a1;
    a0.h[0] = *(const v8bf*)(ap0 + k0 + hf * 8);
    a0.h[1] = *(const v8bf*)(ap0 + k0 + 16 + hf * 8);
    a1.h[0] = *(const v8bf*)(ap1 + k0 + hf * 8);
    a1.h[1] = *(const v8bf*)(ap1 + k0 + 16 + hf * 8);
    union { v16bf v; v8bf h[2]; } bu0, bu1;
    bu0.h[0] = *(const v8bf*)(bp0 + k0 + hf * 8);
    bu0.h[1] = *(const v8bf*)(bp0 + k0 + 16 + hf * 8);
    bu1.h[0] = *(const v8bf*)(bp1 + k0 + hf * 8);
    bu1.h[1] = *(const v8bf*)(bp1 + k0 + 16 + hf * 8);
    v16bf b0 = bu0.v;
    v16bf b1 = bu1.v;
    c00 = __builtin_amdgcn_wmma_f32_16x16x32_bf16(false, a0.v, false, b0,
                                                  (short)0, c00, false, false);
    c01 = __builtin_amdgcn_wmma_f32_16x16x32_bf16(false, a0.v, false, b1,
                                                  (short)0, c01, false, false);
    c10 = __builtin_amdgcn_wmma_f32_16x16x32_bf16(false, a1.v, false, b0,
                                                  (short)0, c10, false, false);
    c11 = __builtin_amdgcn_wmma_f32_16x16x32_bf16(false, a1.v, false, b1,
                                                  (short)0, c11, false, false);
  }
}

// C/D layout: VGPR i -> row m0 + hf*8 + i, col n0 + (lane&15).
__device__ __forceinline__ void store_tile_f32(float* __restrict__ C, int ldc,
                                               int m0, int n0, const v8f& acc,
                                               float bv) {
  const int lane = threadIdx.x & 31;
  const int r = lane & 15, hf = lane >> 4;
  const int col = n0 + r;
#pragma unroll
  for (int i = 0; i < 8; ++i)
    C[(size_t)(m0 + hf * 8 + i) * ldc + col] = acc[i] + bv;
}

__device__ __forceinline__ void store_tile_relu_bf16(__bf16* __restrict__ C, int ldc,
                                                     int m0, int n0, const v8f& acc,
                                                     float bv) {
  const int lane = threadIdx.x & 31;
  const int r = lane & 15, hf = lane >> 4;
  const int col = n0 + r;
#pragma unroll
  for (int i = 0; i < 8; ++i) {
    float v = acc[i] + bv;
    C[(size_t)(m0 + hf * 8 + i) * ldc + col] = (__bf16)(v > 0.f ? v : 0.f);
  }
}

// ------------------------------- pack kernels ------------------------------
__global__ void k_convert_bf16(const float* __restrict__ in,
                               __bf16* __restrict__ out, int n) {
  int i = blockIdx.x * blockDim.x + threadIdx.x;
  if (i < n) out[i] = (__bf16)in[i];
}

// in: rows x cols (f32, row-major) -> out: cols x rows (bf16, row-major)
__global__ void k_transpose_bf16(const float* __restrict__ in,
                                 __bf16* __restrict__ out, int rows, int cols) {
  int i = blockIdx.x * blockDim.x + threadIdx.x;
  if (i < rows * cols) {
    int r = i / cols, c = i - r * cols;
    out[(size_t)c * rows + r] = (__bf16)in[i];
  }
}

// --------------------------- k_proj / q_proj -------------------------------
// M = BS_, N = H_, K = H_ ; 32x32 tiles: 32*8 = 256 per matrix, x2 matrices.
__global__ void k_proj(const __bf16* __restrict__ srcbf,
                       const __bf16* __restrict__ W1t,
                       const __bf16* __restrict__ W2t,
                       const float* __restrict__ b1,
                       const float* __restrict__ b2,
                       float* __restrict__ kproj, float* __restrict__ qproj) {
  const int NT = H_ / 32;                       // 8
  const int TPM = (BS_ / 32) * NT;              // 256
  int wid = (blockIdx.x * blockDim.x + threadIdx.x) >> 5;
  if (wid >= 2 * TPM) return;
  int mat = (wid >= TPM);
  int t   = mat ? wid - TPM : wid;
  int mt = t / NT, nt = t - mt * NT;
  const __bf16* Wt  = mat ? W2t : W1t;
  const float* bias = mat ? b2 : b1;
  float* out        = mat ? qproj : kproj;

  v8f c00 = {}, c01 = {}, c10 = {}, c11 = {};
  wmma_tile32(srcbf + (size_t)(mt * 32) * H_, H_,
              Wt    + (size_t)(nt * 32) * H_, H_, H_, c00, c01, c10, c11);

  int r = threadIdx.x & 15;
  int m0 = mt * 32, n0 = nt * 32;
  float bv0 = bias[n0 + r], bv1 = bias[n0 + 16 + r];
  store_tile_f32(out, H_, m0,      n0,      c00, bv0);
  store_tile_f32(out, H_, m0,      n0 + 16, c01, bv1);
  store_tile_f32(out, H_, m0 + 16, n0,      c10, bv0);
  store_tile_f32(out, H_, m0 + 16, n0 + 16, c11, bv1);
}

// --------------------- fused energy / scores / softmax ---------------------
// One block per (b,t) query row. 8 waves; wave w owns source positions
// s = w, w+8, ... Each lane strides H, reduces with wave32 shfl_xor.
__global__ void k_attn(const float* __restrict__ kproj,
                       const float* __restrict__ qproj,
                       const float* __restrict__ Vvec,
                       const float* __restrict__ bVp,
                       const int* __restrict__ mask,
                       __bf16* __restrict__ attnbf) {
  int b = blockIdx.x / S_, t = blockIdx.x - b * S_;
  int tid = threadIdx.x;
  __shared__ float qrow[H_], vv[H_], sc[S_], red[256];
  qrow[tid] = qproj[(size_t)(b * S_ + t) * H_ + tid];
  vv[tid]   = Vvec[tid];
  __syncthreads();

  int wave = tid >> 5, lane = tid & 31;
  float bV = bVp[0];
  for (int s = wave; s < S_; s += 8) {
    const float* kr = kproj + (size_t)(b * S_ + s) * H_;
    float acc = 0.f;
#pragma unroll
    for (int h = lane; h < H_; h += 32)
      acc += vv[h] * fast_tanh(kr[h] + qrow[h]);
#pragma unroll
    for (int off = 16; off > 0; off >>= 1)
      acc += __shfl_xor(acc, off);
    if (lane == 0)
      sc[s] = (mask[b * S_ + s] == 0) ? -1e10f : (acc + bV);
  }
  __syncthreads();

  // softmax over 512 scores: each thread owns sc[tid] and sc[tid+256]
  float a0 = sc[tid], a1 = sc[tid + 256];
  red[tid] = fmaxf(a0, a1);
  __syncthreads();
  for (int off = 128; off > 0; off >>= 1) {
    if (tid < off) red[tid] = fmaxf(red[tid], red[tid + off]);
    __syncthreads();
  }
  float mx = red[0];
  __syncthreads();
  float e0 = __expf(a0 - mx), e1 = __expf(a1 - mx);
  red[tid] = e0 + e1;
  __syncthreads();
  for (int off = 128; off > 0; off >>= 1) {
    if (tid < off) red[tid] += red[tid + off];
    __syncthreads();
  }
  float rs = 1.0f / red[0];
  __bf16* arow = attnbf + (size_t)(b * S_ + t) * S_;
  arow[tid]       = (__bf16)(e0 * rs);
  arow[tid + 256] = (__bf16)(e1 * rs);
}

// ------------------------------- ctx = attn @ src --------------------------
// Per batch: M = S_, N = H_, K = S_ ; 32x32 tiles: 16*8 = 128 per batch.
__global__ void k_ctx(const __bf16* __restrict__ attnbf,
                      const __bf16* __restrict__ srcT,
                      float* __restrict__ ctx) {
  const int NT = H_ / 32;                       // 8
  const int TPB = (S_ / 32) * NT;               // 128
  int wid = (blockIdx.x * blockDim.x + threadIdx.x) >> 5;
  if (wid >= B_ * TPB) return;
  int b = wid / TPB, t = wid - b * TPB;
  int mt = t / NT, nt = t - mt * NT;

  v8f c00 = {}, c01 = {}, c10 = {}, c11 = {};
  wmma_tile32(attnbf + (size_t)b * S_ * S_ + (size_t)(mt * 32) * S_, S_,
              srcT   + (size_t)b * H_ * S_ + (size_t)(nt * 32) * S_, S_,
              S_, c00, c01, c10, c11);

  float* out = ctx + (size_t)b * S_ * H_;
  int m0 = mt * 32, n0 = nt * 32;
  store_tile_f32(out, H_, m0,      n0,      c00, 0.f);
  store_tile_f32(out, H_, m0,      n0 + 16, c01, 0.f);
  store_tile_f32(out, H_, m0 + 16, n0,      c10, 0.f);
  store_tile_f32(out, H_, m0 + 16, n0 + 16, c11, 0.f);
}

// ----------------------------- residual + LayerNorm ------------------------
__global__ void k_add_ln(const float* __restrict__ a, const float* __restrict__ c,
                         const float* __restrict__ g, const float* __restrict__ be,
                         float* __restrict__ outf, __bf16* __restrict__ outbf) {
  int row = blockIdx.x, tid = threadIdx.x;     // blockDim.x == H_ == 256
  __shared__ float red[H_];
  float x = a[(size_t)row * H_ + tid] + c[(size_t)row * H_ + tid];
  red[tid] = x;
  __syncthreads();
  for (int off = 128; off > 0; off >>= 1) {
    if (tid < off) red[tid] += red[tid + off];
    __syncthreads();
  }
  float mean = red[0] * (1.0f / H_);
  __syncthreads();
  float d = x - mean;
  red[tid] = d * d;
  __syncthreads();
  for (int off = 128; off > 0; off >>= 1) {
    if (tid < off) red[tid] += red[tid + off];
    __syncthreads();
  }
  float rstd = rsqrtf(red[0] * (1.0f / H_) + 1e-5f);
  float y = d * rstd * g[tid] + be[tid];
  if (outf)  outf[(size_t)row * H_ + tid] = y;
  if (outbf) outbf[(size_t)row * H_ + tid] = (__bf16)y;
}

// ------------------------------- FFN GEMMs ---------------------------------
__global__ void k_fc1(const __bf16* __restrict__ xbf,
                      const __bf16* __restrict__ fc1t,
                      const float* __restrict__ fb,
                      __bf16* __restrict__ hbf) {
  const int NT = F_ / 32;                       // 32
  const int TOT = (BS_ / 32) * NT;              // 1024
  int wid = (blockIdx.x * blockDim.x + threadIdx.x) >> 5;
  if (wid >= TOT) return;
  int mt = wid / NT, nt = wid - mt * NT;

  v8f c00 = {}, c01 = {}, c10 = {}, c11 = {};
  wmma_tile32(xbf  + (size_t)(mt * 32) * H_, H_,
              fc1t + (size_t)(nt * 32) * H_, H_, H_, c00, c01, c10, c11);

  int r = threadIdx.x & 15;
  int m0 = mt * 32, n0 = nt * 32;
  float bv0 = fb[n0 + r], bv1 = fb[n0 + 16 + r];
  store_tile_relu_bf16(hbf, F_, m0,      n0,      c00, bv0);
  store_tile_relu_bf16(hbf, F_, m0,      n0 + 16, c01, bv1);
  store_tile_relu_bf16(hbf, F_, m0 + 16, n0,      c10, bv0);
  store_tile_relu_bf16(hbf, F_, m0 + 16, n0 + 16, c11, bv1);
}

__global__ void k_fc2(const __bf16* __restrict__ hbf,
                      const __bf16* __restrict__ fc2t,
                      const float* __restrict__ fb,
                      float* __restrict__ y) {
  const int NT = H_ / 32;                       // 8
  const int TOT = (BS_ / 32) * NT;              // 256
  int wid = (blockIdx.x * blockDim.x + threadIdx.x) >> 5;
  if (wid >= TOT) return;
  int mt = wid / NT, nt = wid - mt * NT;

  v8f c00 = {}, c01 = {}, c10 = {}, c11 = {};
  wmma_tile32(hbf  + (size_t)(mt * 32) * F_, F_,
              fc2t + (size_t)(nt * 32) * F_, F_, F_, c00, c01, c10, c11);

  int r = threadIdx.x & 15;
  int m0 = mt * 32, n0 = nt * 32;
  float bv0 = fb[n0 + r], bv1 = fb[n0 + 16 + r];
  store_tile_f32(y, H_, m0,      n0,      c00, bv0);
  store_tile_f32(y, H_, m0,      n0 + 16, c01, bv1);
  store_tile_f32(y, H_, m0 + 16, n0,      c10, bv0);
  store_tile_f32(y, H_, m0 + 16, n0 + 16, c11, bv1);
}

// ------------------------------- launcher ----------------------------------
extern "C" void kernel_launch(void* const* d_in, const int* in_sizes, int n_in,
                              void* d_out, int out_size, void* d_ws, size_t ws_size,
                              hipStream_t stream) {
  (void)in_sizes; (void)n_in; (void)out_size; (void)ws_size;
  const float* src  = (const float*)d_in[0];
  const int*   mask = (const int*)d_in[1];
  const float* W1   = (const float*)d_in[2];
  const float* b1   = (const float*)d_in[3];
  const float* W2   = (const float*)d_in[4];
  const float* b2   = (const float*)d_in[5];
  const float* Vv   = (const float*)d_in[6];
  const float* bV   = (const float*)d_in[7];
  const float* ln1g = (const float*)d_in[8];
  const float* ln1b = (const float*)d_in[9];
  const float* ln2g = (const float*)d_in[10];
  const float* ln2b = (const float*)d_in[11];
  const float* fc1w = (const float*)d_in[12];
  const float* fc1b = (const float*)d_in[13];
  const float* fc2w = (const float*)d_in[14];
  const float* fc2b = (const float*)d_in[15];
  float* out = (float*)d_out;

  // Bump allocator over workspace (everything L2-resident; ~10.8 MB total).
  char* base = (char*)d_ws;
  size_t off = 0;
  auto alloc = [&](size_t bytes) -> char* {
    char* p = base + off;
    off += (bytes + 255) & ~(size_t)255;
    return p;
  };
  __bf16* srcbf  = (__bf16*)alloc((size_t)BS_ * H_ * 2);
  __bf16* srcT   = (__bf16*)alloc((size_t)B_ * H_ * S_ * 2);
  __bf16* W1t    = (__bf16*)alloc((size_t)H_ * H_ * 2);
  __bf16* W2t    = (__bf16*)alloc((size_t)H_ * H_ * 2);
  __bf16* fc1t   = (__bf16*)alloc((size_t)F_ * H_ * 2);
  __bf16* fc2t   = (__bf16*)alloc((size_t)H_ * F_ * 2);
  float*  kproj  = (float*)alloc((size_t)BS_ * H_ * 4);
  float*  qproj  = (float*)alloc((size_t)BS_ * H_ * 4);
  __bf16* attnbf = (__bf16*)alloc((size_t)BS_ * S_ * 2);
  float*  ctx    = (float*)alloc((size_t)BS_ * H_ * 4);
  float*  xf     = (float*)alloc((size_t)BS_ * H_ * 4);
  __bf16* xbf    = (__bf16*)alloc((size_t)BS_ * H_ * 2);
  __bf16* hbf    = (__bf16*)alloc((size_t)BS_ * F_ * 2);
  float*  yf     = (float*)alloc((size_t)BS_ * H_ * 4);

  // Pack: bf16 conversions + transposes (weights as B^T for contiguous K).
  k_convert_bf16<<<(BS_ * H_ + 255) / 256, 256, 0, stream>>>(src, srcbf, BS_ * H_);
  for (int b = 0; b < B_; ++b)
    k_transpose_bf16<<<(S_ * H_ + 255) / 256, 256, 0, stream>>>(
        src + (size_t)b * S_ * H_, srcT + (size_t)b * H_ * S_, S_, H_);
  k_transpose_bf16<<<(H_ * H_ + 255) / 256, 256, 0, stream>>>(W1, W1t, H_, H_);
  k_transpose_bf16<<<(H_ * H_ + 255) / 256, 256, 0, stream>>>(W2, W2t, H_, H_);
  k_transpose_bf16<<<(H_ * F_ + 255) / 256, 256, 0, stream>>>(fc1w, fc1t, H_, F_);
  k_transpose_bf16<<<(F_ * H_ + 255) / 256, 256, 0, stream>>>(fc2w, fc2t, F_, H_);

  // Projections: 512 wave-tiles (32x32 each) -> 64 blocks of 8 waves.
  k_proj<<<64, 256, 0, stream>>>(srcbf, W1t, W2t, b1, b2, kproj, qproj);

  // Additive-attention energy + softmax (trans-pipe bound, wave32 shuffles).
  k_attn<<<BS_, 256, 0, stream>>>(kproj, qproj, Vv, bV, mask, attnbf);

  // ctx = attn @ src : 256 wave-tiles.
  k_ctx<<<32, 256, 0, stream>>>(attnbf, srcT, ctx);

  // x = LN(src + ctx); keep f32 (residual) and bf16 (fc1 A-operand).
  k_add_ln<<<BS_, 256, 0, stream>>>(src, ctx, ln1g, ln1b, xf, xbf);

  // FFN: 1024 + 256 wave-tiles.
  k_fc1<<<128, 256, 0, stream>>>(xbf, fc1t, fc1b, hbf);
  k_fc2<<<32, 256, 0, stream>>>(hbf, fc2t, fc2b, yf);

  // out = LN(x + y)
  k_add_ln<<<BS_, 256, 0, stream>>>(xf, yf, ln2g, ln2b, out, (__bf16*)nullptr);
}
